// GatedDeltaAttention_29918742184381
// MI455X (gfx1250) — matrix operand order repaired and checked by reading, not compiled, not run on hardware
//
#include <hip/hip_runtime.h>
#include <cmath>

// ---------------- problem constants (match reference) ----------------
#define B_  4
#define T_  1024
#define D_  1024
#define H_  4
#define DK_ 256
#define DV_ 512
#define MT_ (B_ * T_)        // 4096 rows

typedef __bf16 bf16;
typedef __bf16 v16bf __attribute__((ext_vector_type(16)));
typedef float  v8f   __attribute__((ext_vector_type(8)));

// LDS byte offset of a generic pointer into shared memory (AS3 ptrtoint)
__device__ __forceinline__ unsigned lds_offset(const void* p) {
    return (unsigned)(size_t)(__attribute__((address_space(3))) const void*)p;
}

// =====================================================================
// fp32 -> bf16 bulk convert (8 elements / thread, b128 in / b128 out)
// =====================================================================
__launch_bounds__(256)
__global__ void gdn_cvt_bf16(const float* __restrict__ in, bf16* __restrict__ out) {
    size_t i = ((size_t)blockIdx.x * 256 + threadIdx.x) * 8;
    float4 a = *(const float4*)(in + i);
    float4 b = *(const float4*)(in + i + 4);
    union { bf16 h[8]; uint4 u; } o;
    o.h[0] = (bf16)a.x; o.h[1] = (bf16)a.y; o.h[2] = (bf16)a.z; o.h[3] = (bf16)a.w;
    o.h[4] = (bf16)b.x; o.h[5] = (bf16)b.y; o.h[6] = (bf16)b.z; o.h[7] = (bf16)b.w;
    *(uint4*)(out + i) = o.u;
}

// =====================================================================
// WMMA bf16 GEMM:  C[M,N] = A[M,K] @ B[K,N]; A,B bf16 row-major, C fp32.
// Block tile 128x128, K-step 32, 8 waves (4 in M x 2 in N).
// Each wave: 32x64 region = 2x4 accumulators = 8 v_wmma per K-step.
// A tile: async global->LDS copy (global_load_async_to_lds_b128, ASYNCcnt).
// B tile: register-staged with K-pair transpose pack. One barrier / K-step.
// =====================================================================
#define BM 128
#define BN 128
#define BK 32
#define SA 40   // As row stride (bf16): 80B rows -> b128-aligned, conflict-free A frags
#define SB 38   // Bs row stride (bf16): 76B rows -> conflict-free B frag reads

__launch_bounds__(256)
__global__ void gdn_gemm_bf16_wmma(const bf16* __restrict__ A,
                                   const bf16* __restrict__ Bm,
                                   float* __restrict__ C,
                                   int M, int N, int K) {
    __shared__ __align__(16) bf16 As[2][BM][SA];   // A tile, row-major
    __shared__ __align__(16) bf16 Bs[2][BN][SB];   // B tile, transposed: Bs[n][k]

    const int tid  = threadIdx.x;
    const int lane = tid & 31;
    const int wave = tid >> 5;
    const int wm   = wave >> 1;        // 0..3 (M sub-block of 32)
    const int wn   = wave & 1;         // 0..1 (N sub-block of 64)
    const int bm   = blockIdx.y * BM;
    const int bn   = blockIdx.x * BN;

    // fragment addressing (ISA 7.12.2, wave32)
    const int mA  = lane & 15;
    const int khA = (lane >> 4) << 3;  // A: upper lanes hold K+8 group
    const int nB  = lane & 15;
    const int khB = (lane >> 4) << 3;  // B: upper lanes hold K+8 group

    // cooperative load mapping
    const int arow = tid >> 2, ac8 = tid & 3;     // A: rows (arow, arow+64), one b128 each
    const int bkp  = tid >> 4, bn8 = tid & 15;    // B: k-pair 2*bkp, 8 n's from bn8*8

    // ---- A tile: async DMA into LDS (per-lane b128), tracked by ASYNCcnt ----
    auto async_copy_A = [&](int kk, int buf) {
        const bf16* g0 = A + (size_t)(bm + arow)      * K + kk + ac8 * 8;
        const bf16* g1 = A + (size_t)(bm + 64 + arow) * K + kk + ac8 * 8;
        unsigned l0 = lds_offset(&As[buf][arow][ac8 * 8]);
        unsigned l1 = lds_offset(&As[buf][64 + arow][ac8 * 8]);
        asm volatile("global_load_async_to_lds_b128 %0, %1, off"
                     :: "v"(l0), "v"(g0) : "memory");
        asm volatile("global_load_async_to_lds_b128 %0, %1, off"
                     :: "v"(l1), "v"(g1) : "memory");
    };

    // ---- B tile: register-staged, K-pair packed (transposed) ----
    uint4 rb0, rb1;
    auto load_B_regs = [&](int kk) {
        rb0 = *(const uint4*)(Bm + (size_t)(kk + 2 * bkp)     * N + bn + bn8 * 8);
        rb1 = *(const uint4*)(Bm + (size_t)(kk + 2 * bkp + 1) * N + bn + bn8 * 8);
    };
    auto store_B = [&](int buf) {
        unsigned ua[4] = {rb0.x, rb0.y, rb0.z, rb0.w};
        unsigned ub[4] = {rb1.x, rb1.y, rb1.z, rb1.w};
        #pragma unroll
        for (int j = 0; j < 4; ++j) {
            // pack K-pair (k, k+1) per n so fragment reads are contiguous u32s
            unsigned lo = (ua[j] & 0xffffu) | (ub[j] << 16);
            unsigned hi = (ua[j] >> 16)     | (ub[j] & 0xffff0000u);
            *(unsigned*)&Bs[buf][bn8 * 8 + 2 * j][2 * bkp]     = lo;
            *(unsigned*)&Bs[buf][bn8 * 8 + 2 * j + 1][2 * bkp] = hi;
        }
    };

    v8f acc[2][4] = {};
    union Frag { unsigned u[8]; v16bf v; };

    // prologue: stage tile 0 into buffer 0
    async_copy_A(0, 0);
    load_B_regs(0);
    store_B(0);
    int cur = 0;

    for (int k0 = 0; k0 < K; k0 += BK) {
        asm volatile("s_wait_asynccnt 0x0" ::: "memory");  // A tile resident (this wave)
        __syncthreads();                                    // ...and all waves' tiles
        const bool more = (k0 + BK) < K;
        if (more) {
            async_copy_A(k0 + BK, cur ^ 1);   // DMA next A tile during compute
            load_B_regs(k0 + BK);             // next B tile in flight during compute
        }

        Frag fa[2], fb[4];
        #pragma unroll
        for (int im = 0; im < 2; ++im)
            #pragma unroll
            for (int r = 0; r < 8; ++r) {
                int Kp = ((r & 3) << 1) + khA + ((r >> 2) << 4);
                fa[im].u[r] = *(const unsigned*)&As[cur][wm * 32 + im * 16 + mA][Kp];
            }
        #pragma unroll
        for (int in = 0; in < 4; ++in)
            #pragma unroll
            for (int r = 0; r < 8; ++r) {
                int Kq = ((r & 3) << 1) + khB + ((r >> 2) << 4);
                fb[in].u[r] = *(const unsigned*)&Bs[cur][wn * 64 + in * 16 + nB][Kq];
            }
        #pragma unroll
        for (int im = 0; im < 2; ++im)
            #pragma unroll
            for (int in = 0; in < 4; ++in)
                acc[im][in] = __builtin_amdgcn_wmma_f32_16x16x32_bf16(
                    false, fa[im].v, false, fb[in].v, (short)0, acc[im][in], false, false);

        if (more) store_B(cur ^ 1);
        cur ^= 1;
    }

    // C/D layout: VGPR r holds M=r (lanes 0-15) / M=8+r (lanes 16-31), N=lane&15.
    // One base address per output row; N-subtile stores as immediate offsets.
    const int rb = bm + wm * 32 + ((lane >> 4) << 3);
    const int cb = bn + wn * 64 + (lane & 15);
    #pragma unroll
    for (int im = 0; im < 2; ++im)
        #pragma unroll
        for (int r = 0; r < 8; ++r) {
            float* p = C + (size_t)(rb + im * 16 + r) * N + cb;
            p[0]  = acc[im][0][r];
            p[16] = acc[im][1][r];
            p[32] = acc[im][2][r];
            p[48] = acc[im][3][r];
        }
}

// =====================================================================
// beta = sigmoid(x@Wb), g = -exp(A_log)*softplus(x@Wa + dt_bias)
// =====================================================================
__launch_bounds__(256)
__global__ void gdn_beta_g(const float* __restrict__ x,
                           const float* __restrict__ Wb,
                           const float* __restrict__ Wa,
                           const float* __restrict__ A_log,
                           const float* __restrict__ dt_bias,
                           float* __restrict__ beta,
                           float* __restrict__ g) {
    int idx = blockIdx.x * 256 + threadIdx.x;   // B*T*H
    int hh  = idx & (H_ - 1);
    int row = idx >> 2;
    float sb = 0.f, sa = 0.f;
    const float* xr = x + (size_t)row * D_;
    for (int d = 0; d < D_; ++d) {
        float xv = xr[d];
        sb += xv * Wb[d * H_ + hh];
        sa += xv * Wa[d * H_ + hh];
    }
    beta[idx] = 1.f / (1.f + __expf(-sb));
    float z  = sa + dt_bias[hh];
    float sp = (z > 20.f) ? z : log1pf(__expf(z));
    g[idx]   = -__expf(A_log[hh]) * sp;
}

// =====================================================================
// causal depthwise conv (K=4) + SiLU + per-(b,t,h) L2 norm over DK=256.
// =====================================================================
__launch_bounds__(256)
__global__ void gdn_conv_silu_l2(const float* __restrict__ in,   // [B,T,H*DK]
                                 const float* __restrict__ w,    // [H*DK,4]
                                 float* __restrict__ out) {
    const int tid = threadIdx.x;
    const int blk = blockIdx.x;           // (b*T+t)*H + h
    const int h   = blk & (H_ - 1);
    const size_t bt = blk >> 2;           // b*T + t
    const int t   = (int)(bt & (T_ - 1));
    const int c   = h * DK_ + tid;

    float acc = 0.f;
    #pragma unroll
    for (int i = 0; i < 4; ++i) {
        int tt = t - 3 + i;
        if (tt >= 0)
            acc += in[(bt - (size_t)(3 - i)) * (H_ * DK_) + c] * w[c * 4 + i];
    }
    float s = acc / (1.f + __expf(-acc));          // silu

    __shared__ float red[256];
    red[tid] = s * s;
    __syncthreads();
    for (int off = 128; off > 0; off >>= 1) {
        if (tid < off) red[tid] += red[tid + off];
        __syncthreads();
    }
    float inv = rsqrtf(red[0] + 1e-6f);
    out[bt * (H_ * DK_) + c] = s * inv;
}

// causal conv + SiLU, elementwise (for v)
__launch_bounds__(256)
__global__ void gdn_conv_silu(const float* __restrict__ in,  // [B,T,C]
                              const float* __restrict__ w,   // [C,4]
                              float* __restrict__ out, int C) {
    size_t idx = (size_t)blockIdx.x * 256 + threadIdx.x;
    int c = (int)(idx % C);
    size_t bt = idx / C;
    int t = (int)(bt & (T_ - 1));
    float acc = 0.f;
    #pragma unroll
    for (int i = 0; i < 4; ++i) {
        int tt = t - 3 + i;
        if (tt >= 0)
            acc += in[(bt - (size_t)(3 - i)) * C + c] * w[c * 4 + i];
    }
    out[idx] = acc / (1.f + __expf(-acc));
}

// =====================================================================
// gated delta-rule recurrence; state sharded over DV, kept in registers.
// grid = B*H*(DV/64), block = 256. Thread (kseg,dv) owns S[kseg*64..][dv].
// =====================================================================
#define DVC 64
__launch_bounds__(256)
__global__ void gdn_recurrence(const float* __restrict__ q,    // [B,T,H,DK]
                               const float* __restrict__ k,
                               const float* __restrict__ v,    // [B,T,H,DV]
                               const float* __restrict__ g,    // [B,T,H]
                               const float* __restrict__ beta, // [B,T,H]
                               float* __restrict__ o) {        // [B,T,H,DV]
    const int tid   = threadIdx.x;
    const int dv    = tid & (DVC - 1);
    const int kseg  = tid >> 6;                 // 0..3
    const int blk   = blockIdx.x;               // b*H*8 + h*8 + chunk
    const int chunk = blk & 7;
    const int h     = (blk >> 3) & (H_ - 1);
    const int b     = blk >> 5;

    __shared__ float ksh[DK_], qsh[DK_], red1[256], red2[256];

    float S[64];
    #pragma unroll
    for (int i = 0; i < 64; ++i) S[i] = 0.f;

    const float scale = 0.0625f;                // DK^-0.5

    for (int t = 0; t < T_; ++t) {
        size_t base = (((size_t)b * T_ + t) * H_ + h);
        ksh[tid] = k[base * DK_ + tid];
        qsh[tid] = q[base * DK_ + tid];
        float gt = g[base];
        float bt = beta[base];
        float vt = v[base * DV_ + chunk * DVC + dv];
        __syncthreads();

        float eg  = __expf(gt);
        float kvp = 0.f;
        const float* kp = &ksh[kseg * 64];
        #pragma unroll
        for (int i = 0; i < 64; ++i) { S[i] *= eg; kvp += kp[i] * S[i]; }
        red1[tid] = kvp;
        __syncthreads();

        float kv = red1[dv] + red1[dv + 64] + red1[dv + 128] + red1[dv + 192];
        float delta = (vt - kv) * bt;

        const float* qp = &qsh[kseg * 64];
        float op = 0.f;
        #pragma unroll
        for (int i = 0; i < 64; ++i) { S[i] += kp[i] * delta; op += qp[i] * S[i]; }
        red2[tid] = op;
        __syncthreads();

        if (kseg == 0) {
            float ov = red2[dv] + red2[dv + 64] + red2[dv + 128] + red2[dv + 192];
            o[base * DV_ + chunk * DVC + dv] = ov * scale;
        }
        __syncthreads();
    }
}

// =====================================================================
// gated RMSNorm -> bf16 output (feeds final WMMA GEMM directly)
// =====================================================================
__launch_bounds__(256)
__global__ void gdn_gated_rmsnorm(const float* __restrict__ o_raw, // [B,T,H,DV]
                                  const float* __restrict__ gate,  // [B,T,H*DV]
                                  const float* __restrict__ norm_w,// [DV]
                                  bf16* __restrict__ out) {        // [B*T,H*DV] bf16
    const int blk = blockIdx.x;                 // (b*T+t)*H + h
    const int h   = blk & (H_ - 1);
    const size_t row = blk >> 2;
    const int tid = threadIdx.x;

    __shared__ float red[256];
    size_t obase = (size_t)blk * DV_;
    float v0 = o_raw[obase + tid];
    float v1 = o_raw[obase + 256 + tid];
    red[tid] = v0 * v0 + v1 * v1;
    __syncthreads();
    for (int off = 128; off > 0; off >>= 1) {
        if (tid < off) red[tid] += red[tid + off];
        __syncthreads();
    }
    float inv = rsqrtf(red[0] * (1.0f / DV_) + 1e-5f);

    size_t gbase = row * (H_ * DV_) + (size_t)h * DV_;
    float g0 = gate[gbase + tid];
    float g1 = gate[gbase + 256 + tid];
    out[gbase + tid]       = (bf16)(v0 * inv * norm_w[tid]       * (g0 / (1.f + __expf(-g0))));
    out[gbase + 256 + tid] = (bf16)(v1 * inv * norm_w[256 + tid] * (g1 / (1.f + __expf(-g1))));
}

// =====================================================================
extern "C" void kernel_launch(void* const* d_in, const int* in_sizes, int n_in,
                              void* d_out, int out_size, void* d_ws, size_t ws_size,
                              hipStream_t stream) {
    const float* x       = (const float*)d_in[0];
    const float* Wq      = (const float*)d_in[1];
    const float* Wk      = (const float*)d_in[2];
    const float* Wv      = (const float*)d_in[3];
    const float* conv_qw = (const float*)d_in[4];
    const float* conv_kw = (const float*)d_in[5];
    const float* conv_vw = (const float*)d_in[6];
    const float* Wb      = (const float*)d_in[7];
    const float* Wa      = (const float*)d_in[8];
    const float* A_log   = (const float*)d_in[9];
    const float* dt_bias = (const float*)d_in[10];
    const float* Wg      = (const float*)d_in[11];
    const float* norm_w  = (const float*)d_in[12];
    const float* Wo      = (const float*)d_in[13];
    float* out = (float*)d_out;
    (void)in_sizes; (void)n_in; (void)out_size; (void)ws_size;

    char* wsb = (char*)d_ws;
    size_t off = 0;
    auto alloc = [&](size_t bytes) {
        void* p = wsb + off;
        off += (bytes + 255) & ~(size_t)255;
        return p;
    };

    // bf16 operands for WMMA GEMMs
    bf16* xbf  = (bf16*)alloc((size_t)MT_ * D_ * 2);
    bf16* Wqbf = (bf16*)alloc((size_t)D_ * (H_ * DK_) * 2);
    bf16* Wkbf = (bf16*)alloc((size_t)D_ * (H_ * DK_) * 2);
    bf16* Wvbf = (bf16*)alloc((size_t)D_ * (H_ * DV_) * 2);
    bf16* Wgbf = (bf16*)alloc((size_t)D_ * (H_ * DV_) * 2);
    bf16* Wobf = (bf16*)alloc((size_t)(H_ * DV_) * D_ * 2);
    bf16* onbf = (bf16*)alloc((size_t)MT_ * (H_ * DV_) * 2);
    // fp32 intermediates
    float* qpre  = (float*)alloc((size_t)MT_ * (H_ * DK_) * 4);
    float* kpre  = (float*)alloc((size_t)MT_ * (H_ * DK_) * 4);
    float* vpre  = (float*)alloc((size_t)MT_ * (H_ * DV_) * 4);
    float* gateb = (float*)alloc((size_t)MT_ * (H_ * DV_) * 4);
    float* qbuf  = (float*)alloc((size_t)MT_ * (H_ * DK_) * 4);
    float* kbuf  = (float*)alloc((size_t)MT_ * (H_ * DK_) * 4);
    float* vbuf  = (float*)alloc((size_t)MT_ * (H_ * DV_) * 4);
    float* betab = (float*)alloc((size_t)MT_ * H_ * 4);
    float* gbuf  = (float*)alloc((size_t)MT_ * H_ * 4);
    float* oraw  = (float*)alloc((size_t)MT_ * (H_ * DV_) * 4);

    dim3 blk256(256);
    auto cvt = [&](const float* src, bf16* dst, size_t n) {
        gdn_cvt_bf16<<<dim3((unsigned)(n / 2048)), blk256, 0, stream>>>(src, dst);
    };

    // one-time bf16 conversions
    cvt(x,  xbf,  (size_t)MT_ * D_);
    cvt(Wq, Wqbf, (size_t)D_ * (H_ * DK_));
    cvt(Wk, Wkbf, (size_t)D_ * (H_ * DK_));
    cvt(Wv, Wvbf, (size_t)D_ * (H_ * DV_));
    cvt(Wg, Wgbf, (size_t)D_ * (H_ * DV_));
    cvt(Wo, Wobf, (size_t)(H_ * DV_) * D_);

    // projections (WMMA GEMMs)
    dim3 gQK(1024 / BN, MT_ / BM);
    dim3 gVG(2048 / BN, MT_ / BM);
    gdn_gemm_bf16_wmma<<<gQK, blk256, 0, stream>>>(xbf, Wqbf, qpre, MT_, H_ * DK_, D_);
    gdn_gemm_bf16_wmma<<<gQK, blk256, 0, stream>>>(xbf, Wkbf, kpre, MT_, H_ * DK_, D_);
    gdn_gemm_bf16_wmma<<<gVG, blk256, 0, stream>>>(xbf, Wvbf, vpre, MT_, H_ * DV_, D_);
    gdn_gemm_bf16_wmma<<<gVG, blk256, 0, stream>>>(xbf, Wgbf, gateb, MT_, H_ * DV_, D_);

    // beta / g
    gdn_beta_g<<<dim3((MT_ * H_) / 256), blk256, 0, stream>>>(x, Wb, Wa, A_log, dt_bias,
                                                              betab, gbuf);

    // conv + silu (+ l2norm for q,k)
    gdn_conv_silu_l2<<<dim3(MT_ * H_), blk256, 0, stream>>>(qpre, conv_qw, qbuf);
    gdn_conv_silu_l2<<<dim3(MT_ * H_), blk256, 0, stream>>>(kpre, conv_kw, kbuf);
    gdn_conv_silu<<<dim3((unsigned)((size_t)MT_ * (H_ * DV_) / 256)), blk256, 0, stream>>>(
        vpre, conv_vw, vbuf, H_ * DV_);

    // gated delta-rule recurrence
    gdn_recurrence<<<dim3(B_ * H_ * (DV_ / DVC)), blk256, 0, stream>>>(
        qbuf, kbuf, vbuf, gbuf, betab, oraw);

    // gated RMSNorm -> bf16
    gdn_gated_rmsnorm<<<dim3(MT_ * H_), blk256, 0, stream>>>(oraw, gateb, norm_w, onbf);

    // output projection (WMMA GEMM)
    gdn_gemm_bf16_wmma<<<dim3(1024 / BN, MT_ / BM), blk256, 0, stream>>>(
        onbf, Wobf, out, MT_, D_, H_ * DV_);
}
